// GRUModel_27462020891343
// MI455X (gfx1250) — matrix operand order repaired and checked by reading, not compiled, not run on hardware
//
#include <hip/hip_runtime.h>

typedef __attribute__((ext_vector_type(16))) _Float16 v16h;
typedef __attribute__((ext_vector_type(8)))  float    v8f;
typedef __attribute__((ext_vector_type(8)))  unsigned int v8u;
typedef __attribute__((ext_vector_type(4)))  unsigned int v4u;

#define WMMA_F16(A, B, C) \
  __builtin_amdgcn_wmma_f32_16x16x32_f16(false, (A), false, (B), (short)0, (C), false, false)

constexpr int B_ = 128, S_ = 144, T_ = 50, I_ = 10, H_ = 64;
constexpr int BM = 32;    // batch rows per block
constexpr int XS = 32;    // sX row stride (K padded 10 -> 32), halves
constexpr int HS = 72;    // sH row stride, halves (padded)
constexpr int FS = 132;   // sF row stride, floats (padded)
constexpr int STI = S_ * T_ * I_;
constexpr int TI  = T_ * I_;

// ---- fast activations from CDNA5 trans ops (no divides, no branches) ----
// NOTE: __has_builtin(__builtin_amdgcn_*) is false in the HOST pass, so the
// fallbacks must be side-neutral Clang builtins / plain arithmetic.
__device__ __forceinline__ float frcp(float x) {
#if __has_builtin(__builtin_amdgcn_rcpf)
  return __builtin_amdgcn_rcpf(x);          // V_RCP_F32
#else
  return 1.0f / x;
#endif
}
__device__ __forceinline__ float fexp2(float x) {
#if __has_builtin(__builtin_amdgcn_exp2f)
  return __builtin_amdgcn_exp2f(x);         // V_EXP_F32
#else
  return __builtin_exp2f(x);                // portable clang builtin -> llvm.exp2
#endif
}
__device__ __forceinline__ float sigf(float x) {
  // 1/(1+e^-x) = rcp(1 + 2^(-log2e * x))
  return frcp(1.0f + fexp2(-1.44269504088896340736f * x));
}
__device__ __forceinline__ float tanh_fast(float x) {
#if __has_builtin(__builtin_amdgcn_tanhf)
  return __builtin_amdgcn_tanhf(x);         // V_TANH_F32
#else
  float e = fexp2(-2.88539008177792681472f * x);  // e^(-2x)
  return (1.0f - e) * frcp(1.0f + e);
#endif
}

__device__ __forceinline__ v8f bcast(float v) { return (v8f){v, v, v, v, v, v, v, v}; }

// A-fragment (16-bit A 16x32 layout): halves 0..15 = K 2*khi..2*khi+15 (contiguous, matches B)
__device__ __forceinline__ v16h loadA(const _Float16* rowp, int khi) {
  v4u lo = *(const v4u*)(rowp + 2 * khi);
  v4u hi = *(const v4u*)(rowp + 2 * khi + 8);
  v8u u;
  u[0] = lo[0]; u[1] = lo[1]; u[2] = lo[2]; u[3] = lo[3];
  u[4] = hi[0]; u[5] = hi[1]; u[6] = hi[2]; u[7] = hi[3];
  return __builtin_bit_cast(v16h, u);
}

__global__ __launch_bounds__(256) void gru_fused(
    const float* __restrict__ x,
    const float* __restrict__ WihF, const float* __restrict__ WhhF,
    const float* __restrict__ bihF, const float* __restrict__ bhhF,
    const float* __restrict__ WihB,
    const float* __restrict__ bihB, const float* __restrict__ bhhB,
    const float* __restrict__ Wfc,  const float* __restrict__ bfc,
    float* __restrict__ out)
{
  __shared__ __align__(16) _Float16 sX[BM * XS];   // x_t, f16, K padded to 32
  __shared__ __align__(16) _Float16 sH[BM * HS];   // h, f16 copy for WMMA A
  __shared__ __align__(16) float    sF[BM * FS];   // [hf | hb] f32 for FC tail

  const int t   = blockIdx.x >> 2;            // turbine
  const int b0  = (blockIdx.x & 3) * BM;      // batch slice
  const int tid = threadIdx.x;
  const int w = tid >> 5, lane = tid & 31;
  const int mi = w >> 2, q = w & 3;           // batch tile / gate-col group
  const int ln = lane & 15, lh = lane >> 4;
  const int khi = lh << 3;                    // A-frag K base
  const int kb  = lh << 4;                    // B-frag K base

  const float* wih = WihF + (size_t)t * (3 * H_ * I_);
  const float* whh = WhhF + (size_t)t * (3 * H_ * H_);

  // ---- preload weight B-fragments into registers (held across the scan) ----
  v16h Bx[3];         // Wih (K=10 padded to 32)
  v16h Bh[3][2];      // Whh, two K=32 halves
  float brz[2];       // bih+bhh for r,z (merged accumulator seed)
  float bin_ = 0.0f, bhn_ = 0.0f;   // n-gate biases kept separate
#pragma unroll
  for (int g = 0; g < 3; ++g) {
    const int j = g * 64 + q * 16 + ln;
    float bi = bihF[t * 192 + j], bh = bhhF[t * 192 + j];
    if (g < 2) brz[g] = bi + bh; else { bin_ = bi; bhn_ = bh; }
    v16h bx;
#pragma unroll
    for (int e = 0; e < 16; ++e) {
      int k = kb + e;
      bx[e] = (k < I_) ? (_Float16)wih[j * I_ + k] : (_Float16)0.0f;
    }
    Bx[g] = bx;
#pragma unroll
    for (int kh = 0; kh < 2; ++kh) {
      const float4* wp = (const float4*)(whh + j * H_ + kh * 32 + kb);
      v16h bhv;
#pragma unroll
      for (int u4 = 0; u4 < 4; ++u4) {
        float4 f = wp[u4];
        bhv[4 * u4 + 0] = (_Float16)f.x; bhv[4 * u4 + 1] = (_Float16)f.y;
        bhv[4 * u4 + 2] = (_Float16)f.z; bhv[4 * u4 + 3] = (_Float16)f.w;
      }
      Bh[g][kh] = bhv;
    }
  }

  // ---- x staging: fixed per-thread (row,k) slots, k<10 only; pads written once ----
  // slot0: e = tid (all 256 threads), slot1: e = tid+256 (tid<64); e in [0,320)
  int e0 = tid, bl0 = e0 / 10, k0 = e0 - bl0 * 10;
  const float* p0 = x + (size_t)(b0 + bl0) * STI + t * I_ + k0;
  const int la0 = bl0 * XS + k0;
  int e1 = tid + 256, bl1 = e1 / 10, k1 = e1 - bl1 * 10;
  const float* p1 = x + (size_t)(b0 + (bl1 < BM ? bl1 : 0)) * STI + t * I_ + k1;
  const int la1 = bl1 * XS + k1;
  const bool has1 = (e1 < BM * I_);

  // zero pads of sX (k >= 10) and all of sH; then stage x(s=0)
  for (int e = tid; e < BM * XS; e += 256) {
    if ((e & 31) >= I_) sX[e] = (_Float16)0.0f;
  }
  for (int e = tid; e < BM * HS; e += 256) sH[e] = (_Float16)0.0f;
  sX[la0] = (_Float16)p0[0];
  if (has1) sX[la1] = (_Float16)p1[0];
  p0 += TI; p1 += TI;

  v8f h = {};                       // hidden-state patch lives in registers
  const int am = mi * 16 + ln;      // A-frag row for this lane
  __syncthreads();

  // ---- sequential forward scan ----
  for (int s = 0; s < S_; ++s) {
    v16h Ax  = loadA(&sX[am * XS], khi);
    v16h Ah0 = loadA(&sH[am * HS], khi);
    v16h Ah1 = loadA(&sH[am * HS + 32], khi);

    // r,z gates: gi+gh merged into one accumulator (seeded with bih+bhh)
    v8f Drz[2];
#pragma unroll
    for (int g = 0; g < 2; ++g) {
      v8f d = bcast(brz[g]);
      d = WMMA_F16(Ax,  Bx[g],    d);
      d = WMMA_F16(Ah0, Bh[g][0], d);
      d = WMMA_F16(Ah1, Bh[g][1], d);
      Drz[g] = d;
    }
    // n gate: gi and gh kept separate (n = tanh(gi_n + r*gh_n))
    v8f Dxn = bcast(bin_);
    Dxn = WMMA_F16(Ax, Bx[2], Dxn);
    v8f Dhn = bcast(bhn_);
    Dhn = WMMA_F16(Ah0, Bh[2][0], Dhn);
    Dhn = WMMA_F16(Ah1, Bh[2][1], Dhn);

    __syncthreads();  // all sH/sX reads complete before overwrite

#pragma unroll
    for (int r = 0; r < 8; ++r) {
      float rr = sigf(Drz[0][r]);
      float zz = sigf(Drz[1][r]);
      float nn = tanh_fast(fmaf(rr, Dhn[r], Dxn[r]));
      float hp = fmaf(zz, h[r] - nn, nn);   // n + z*(h-n)
      h[r] = hp;
      sH[(mi * 16 + r + khi) * HS + (q * 16 + ln)] = (_Float16)hp;
    }

    if (s + 1 < S_) {               // restage: 1-2 loads per thread, pointer bump
      sX[la0] = (_Float16)p0[0];
      if (has1) sX[la1] = (_Float16)p1[0];
      p0 += TI; p1 += TI;
    }
    __syncthreads();
  }

  // ---- backward cell at s = S-1 (h0 = 0 => gh = bhh_b, no Whh_b GEMM) ----
  const float* wihb = WihB + (size_t)t * (3 * H_ * I_);
  v8f Db[3]; float bhb[3];
  {
    v16h Axl = loadA(&sX[am * XS], khi);   // sX still holds x(S-1)
#pragma unroll
    for (int g = 0; g < 3; ++g) {
      const int j = g * 64 + q * 16 + ln;
      bhb[g] = bhhB[t * 192 + j];
      v16h bb;
#pragma unroll
      for (int e = 0; e < 16; ++e) {
        int k = kb + e;
        bb[e] = (k < I_) ? (_Float16)wihb[j * I_ + k] : (_Float16)0.0f;
      }
      v8f d = bcast(bihB[t * 192 + j]);
      d = WMMA_F16(Axl, bb, d);
      Db[g] = d;
    }
  }
#pragma unroll
  for (int r = 0; r < 8; ++r) {
    float rr = sigf(Db[0][r] + bhb[0]);
    float zz = sigf(Db[1][r] + bhb[1]);
    float nn = tanh_fast(fmaf(rr, bhb[2], Db[2][r]));
    float hb = (1.0f - zz) * nn;
    int m = mi * 16 + r + khi, c = q * 16 + ln;
    sF[m * FS + c]      = h[r];   // hf
    sF[m * FS + 64 + c] = hb;     // hb
  }
  __syncthreads();

  // ---- FC tail: out[b,0,t] = [hf|hb] . Wfc[t,0,:] + bfc[t] ----
  if (tid < BM) {
    const float* wf = Wfc + (size_t)t * 128;
    float acc = bfc[t];
#pragma unroll 8
    for (int j2 = 0; j2 < 128; ++j2) acc += sF[tid * FS + j2] * wf[j2];
    out[(size_t)(b0 + tid) * T_ + t] = acc;
  }
}

extern "C" void kernel_launch(void* const* d_in, const int* in_sizes, int n_in,
                              void* d_out, int out_size, void* d_ws, size_t ws_size,
                              hipStream_t stream) {
  const float* x    = (const float*)d_in[0];
  const float* WihF = (const float*)d_in[1];
  const float* WhhF = (const float*)d_in[2];
  const float* bihF = (const float*)d_in[3];
  const float* bhhF = (const float*)d_in[4];
  const float* WihB = (const float*)d_in[5];
  // d_in[6] = Whh_b: unused (backward h0 == 0 so gh == bhh_b)
  const float* bihB = (const float*)d_in[7];
  const float* bhhB = (const float*)d_in[8];
  const float* Wfc  = (const float*)d_in[9];
  const float* bfc  = (const float*)d_in[10];
  float* out = (float*)d_out;

  gru_fused<<<dim3(T_ * (B_ / BM)), 256, 0, stream>>>(
      x, WihF, WhhF, bihF, bhhF, WihB, bihB, bhhB, Wfc, bfc, out);
}
